// MoELayer_76596446757256
// MI455X (gfx1250) — matrix operand order repaired and checked by reading, not compiled, not run on hardware
//
#include <hip/hip_runtime.h>
#include <math.h>
#include <stdint.h>

// ---------------------------------------------------------------------------
// Problem constants (B=2, T=2048, C=1024, E=8, H=4096, top-2)
// ---------------------------------------------------------------------------
constexpr int NTOK    = 4096;            // B*T tokens
constexpr int CDIM    = 1024;
constexpr int EXP     = 8;
constexpr int HDIM    = 4096;
constexpr int TOPK    = 2;
constexpr int NASSIGN = NTOK * TOPK;     // 8192 routed assignments
constexpr int BLKM    = 128;             // M tile (rows of sorted assignments)
constexpr int MAXT    = NASSIGN / BLKM + EXP;  // worst-case M tiles = 72
constexpr int PADMAX  = MAXT * BLKM;     // padded sorted-list capacity = 9216

typedef __attribute__((ext_vector_type(16))) __bf16 bf16x16;
typedef __attribute__((ext_vector_type(8)))  __bf16 bf16x8;
typedef __attribute__((ext_vector_type(8)))  float  f32x8;

// ---------------------------------------------------------------------------
// helpers
// ---------------------------------------------------------------------------
__device__ __forceinline__ float gelu_new(float x) {
  // 0.5*x*(1+tanh(c*(x+0.044715x^3))) == x * sigmoid(2c*(x+0.044715x^3))
  const float two_c = 2.0f * 0.7978845608028654f;
  const float t = two_c * (x + 0.044715f * x * x * x);
  return x / (1.0f + __expf(-t));        // branch-free, saturates correctly
}

// CDNA5 16-bit A fragment (16x32): lane half h holds K in two contiguous runs
// [8h .. 8h+7] and [16+8h .. 16+8h+7]  ->  two 16B row-major loads.
__device__ __forceinline__ bf16x16 load_a_frag(const __bf16* row, int k0, int half) {
  union { bf16x16 v; bf16x8 h[2]; } u;
  u.h[0] = *(const bf16x8*)(row + k0 + 8 * half);
  u.h[1] = *(const bf16x8*)(row + k0 + 16 + 8 * half);
  return u.v;
}

// Async copy of one 16B chunk global -> LDS (ASYNCcnt-tracked).
__device__ __forceinline__ void async_g2l_b128(uint32_t lds_addr, uint64_t gaddr) {
  asm volatile("global_load_async_to_lds_b128 %0, %1, off"
               :: "v"(lds_addr), "v"(gaddr) : "memory");
}
__device__ __forceinline__ void wait_async_le2() {
  asm volatile("s_wait_asynccnt 0x2" ::: "memory");
}
__device__ __forceinline__ void wait_async_0() {
  asm volatile("s_wait_asynccnt 0x0" ::: "memory");
}

// ---------------------------------------------------------------------------
// 1) Router: one wave (32 lanes) per token.
// ---------------------------------------------------------------------------
__global__ __launch_bounds__(256)
void k_router(const float* __restrict__ x, const float* __restrict__ rw,
              int* __restrict__ topk_idx, float* __restrict__ topk_w) {
  const int wave = threadIdx.x >> 5;
  const int lane = threadIdx.x & 31;
  const int tok  = blockIdx.x * 8 + wave;
  const float* xr = x + (size_t)tok * CDIM;

  float acc[EXP];
#pragma unroll
  for (int e = 0; e < EXP; ++e) acc[e] = 0.0f;

  for (int c0 = lane * 4; c0 < CDIM; c0 += 128) {
    float4 xv = *(const float4*)(xr + c0);
#pragma unroll
    for (int e = 0; e < EXP; ++e) {
      float4 wv = *(const float4*)(rw + e * CDIM + c0);
      acc[e] += xv.x * wv.x + xv.y * wv.y + xv.z * wv.z + xv.w * wv.w;
    }
  }
#pragma unroll
  for (int e = 0; e < EXP; ++e)
#pragma unroll
    for (int off = 16; off; off >>= 1)
      acc[e] += __shfl_xor(acc[e], off, 32);

  if (lane == 0) {
    int i1 = 0;
    for (int e = 1; e < EXP; ++e) if (acc[e] > acc[i1]) i1 = e;
    int i2 = -1;
    for (int e = 0; e < EXP; ++e) {
      if (e == i1) continue;
      if (i2 < 0 || acc[e] > acc[i2]) i2 = e;
    }
    const float w1 = 1.0f / (1.0f + expf(acc[i2] - acc[i1]));
    topk_idx[tok * 2 + 0] = i1;  topk_w[tok * 2 + 0] = w1;
    topk_idx[tok * 2 + 1] = i2;  topk_w[tok * 2 + 1] = 1.0f - w1;
  }
}

// ---------------------------------------------------------------------------
// 2) Routing bookkeeping
// ---------------------------------------------------------------------------
__global__ void k_init(int* __restrict__ sorted_ids, int* __restrict__ counts) {
  const int i = blockIdx.x * blockDim.x + threadIdx.x;
  if (i < PADMAX) sorted_ids[i] = -1;
  if (i < EXP) counts[i] = 0;
}

__global__ void k_count(const int* __restrict__ topk_idx, int* __restrict__ counts) {
  const int a = blockIdx.x * blockDim.x + threadIdx.x;
  if (a < NASSIGN) atomicAdd(&counts[topk_idx[a]], 1);
}

__global__ void k_meta(const int* __restrict__ counts, int* __restrict__ fill,
                       int* __restrict__ tile_expert, int* __restrict__ num_tiles) {
  if (threadIdx.x != 0 || blockIdx.x != 0) return;
  int base = 0, t = 0;
  for (int e = 0; e < EXP; ++e) {
    fill[e] = base;
    const int nt = (counts[e] + BLKM - 1) / BLKM;
    for (int i = 0; i < nt; ++i) tile_expert[t++] = e;
    base += nt * BLKM;              // block-aligned segment start per expert
  }
  *num_tiles = t;
}

__global__ void k_scatter(const int* __restrict__ topk_idx, const float* __restrict__ topk_w,
                          int* __restrict__ fill, int* __restrict__ sorted_ids,
                          float* __restrict__ sorted_w) {
  const int a = blockIdx.x * blockDim.x + threadIdx.x;
  if (a >= NASSIGN) return;
  const int e = topk_idx[a];
  const int pos = atomicAdd(&fill[e], 1);
  sorted_ids[pos] = a;              // assignment id: token*2 + k
  sorted_w[pos]   = topk_w[a];
}

// ---------------------------------------------------------------------------
// 3) fp32 -> bf16 conversion / B-fragment swizzle
//    Fragment layout grouped by 64-col block so the 4 fragments of a K-step
//    are one contiguous 4KB chunk (single async-staged block):
//    dst frag = (((e*(NC/64) + nb)*(K/32) + kt)*4 + ts)  with nt = nb*4+ts
//    within fragment: lane = n + 16*half holds 16 bf16, K = kt*32+(idx<8 ? 8*half+idx : 16+8*half+idx-8)
// ---------------------------------------------------------------------------
__global__ void k_cvt_x(const float* __restrict__ x, __bf16* __restrict__ xb) {
  const int i = (blockIdx.x * blockDim.x + threadIdx.x) * 4;
  float4 v = *(const float4*)(x + i);
  xb[i + 0] = (__bf16)v.x; xb[i + 1] = (__bf16)v.y;
  xb[i + 2] = (__bf16)v.z; xb[i + 3] = (__bf16)v.w;
}

__global__ __launch_bounds__(256)
void k_cvt_w(const float* __restrict__ w, __bf16* __restrict__ out, int K, int NC) {
  const int gw   = (blockIdx.x * blockDim.x + threadIdx.x) >> 5;  // fragment id
  const int lane = threadIdx.x & 31;
  const int half = lane >> 4, ln = lane & 15;
  const int KT = K / 32, NT = NC / 16;
  if (gw >= EXP * NT * KT) return;
  const int kt = gw % KT;
  const int nt = (gw / KT) % NT;
  const int e  = gw / (KT * NT);
  const float* we = w + (size_t)e * K * NC;
  __bf16 tmp[16];
#pragma unroll
  for (int idx = 0; idx < 16; ++idx) {
    const int c = kt * 32 + (idx < 8 ? 8 * half + idx : 16 + 8 * half + (idx - 8));   // K index
    const int n = nt * 16 + ln;                // column index
    tmp[idx] = (__bf16)we[(size_t)c * NC + n];
  }
  const int nb = nt >> 2, ts = nt & 3;
  __bf16* o = out + ((((size_t)e * (NC / 64) + nb) * KT + kt) * 4 + ts) * 512 + lane * 16;
#pragma unroll
  for (int idx = 0; idx < 16; ++idx) o[idx] = tmp[idx];
}

// ---------------------------------------------------------------------------
// 4) GEMM1: h = gelu(x_gathered @ w_fc[e] + b_fc[e]) -> bf16 h_ws (sorted order)
//    128 threads = 4 waves; wave owns 32 rows x 64 cols (8 WMMA accums).
//    B block (4KB per K-step) double-buffered in LDS via async-to-LDS copies.
// ---------------------------------------------------------------------------
__global__ __launch_bounds__(128)
void k_gemm1(const __bf16* __restrict__ xb, const __bf16* __restrict__ wfc,
             const float* __restrict__ bfc, const int* __restrict__ sorted_ids,
             const int* __restrict__ tile_expert, const int* __restrict__ num_tiles,
             __bf16* __restrict__ h_ws) {
  __shared__ __attribute__((aligned(16))) __bf16 smem[2][2048];   // 2 x 4KB
  const int mt = blockIdx.x;
  if (mt >= *num_tiles) return;
  const int e    = tile_expert[mt];
  const int row0 = mt * BLKM;
  const int wave = threadIdx.x >> 5;
  const int lane = threadIdx.x & 31;
  const int half = lane >> 4, ln = lane & 15;
  const int nb   = blockIdx.y;                 // 64-col block of H

  // A rows (gathered token rows), two 16-row strips per wave
  const int r0 = row0 + wave * 32 + ln;
  const int a0 = sorted_ids[r0];
  const int a1 = sorted_ids[r0 + 16];
  const __bf16* arow0 = xb + (size_t)(a0 < 0 ? 0 : (a0 >> 1)) * CDIM;
  const __bf16* arow1 = xb + (size_t)(a1 < 0 ? 0 : (a1 >> 1)) * CDIM;

  constexpr int KT = CDIM / 32;
  const __bf16* bsrc = wfc + ((size_t)e * (HDIM / 64) + nb) * KT * 2048;
  const uint32_t lds0 = (uint32_t)(uintptr_t)&smem[0][0];
  const uint32_t lds1 = (uint32_t)(uintptr_t)&smem[1][0];
  const uint32_t loff = threadIdx.x * 32;      // 128 thr x 32B = 4KB

  auto stage = [&](int kt, int buf) {
    const uint64_t g = (uint64_t)(uintptr_t)(bsrc + (size_t)kt * 2048) + loff;
    const uint32_t l = (buf ? lds1 : lds0) + loff;
    async_g2l_b128(l, g);
    async_g2l_b128(l + 16, g + 16);
  };

  f32x8 acc[8] = {};
  stage(0, 0);
  for (int kt = 0; kt < KT; ++kt) {
    const int buf = kt & 1;
    if (kt + 1 < KT) { stage(kt + 1, buf ^ 1); wait_async_le2(); }
    else             { wait_async_0(); }
    __syncthreads();                           // all waves' copies landed
    const bf16x16 af0 = load_a_frag(arow0, kt * 32, half);
    const bf16x16 af1 = load_a_frag(arow1, kt * 32, half);
#pragma unroll
    for (int t = 0; t < 4; ++t) {
      const bf16x16 b = *(const bf16x16*)&smem[buf][t * 512 + lane * 16];
      acc[t]     = __builtin_amdgcn_wmma_f32_16x16x32_bf16(false, af0, false, b,
                                                           (short)0, acc[t], false, false);
      acc[4 + t] = __builtin_amdgcn_wmma_f32_16x16x32_bf16(false, af1, false, b,
                                                           (short)0, acc[4 + t], false, false);
    }
    __syncthreads();                           // reads done before overwrite
  }

#pragma unroll
  for (int t = 0; t < 4; ++t) {
    const int hcol = nb * 64 + t * 16 + ln;
    const float bias = bfc[e * HDIM + hcol];
#pragma unroll
    for (int s = 0; s < 2; ++s)
#pragma unroll
      for (int i = 0; i < 8; ++i) {
        const int slot = row0 + wave * 32 + s * 16 + i + 8 * half;
        h_ws[(size_t)slot * HDIM + hcol] = (__bf16)gelu_new(acc[4 * s + t][i] + bias);
      }
  }
}

// ---------------------------------------------------------------------------
// 5) GEMM2: out_partial[assignment] = weight * (h @ w_proj[e] + b_proj[e])
// ---------------------------------------------------------------------------
__global__ __launch_bounds__(128)
void k_gemm2(const __bf16* __restrict__ h_ws, const __bf16* __restrict__ wpj,
             const float* __restrict__ bpj, const int* __restrict__ sorted_ids,
             const float* __restrict__ sorted_w, const int* __restrict__ tile_expert,
             const int* __restrict__ num_tiles, float* __restrict__ out_partial) {
  __shared__ __attribute__((aligned(16))) __bf16 smem[2][2048];
  const int mt = blockIdx.x;
  if (mt >= *num_tiles) return;
  const int e    = tile_expert[mt];
  const int row0 = mt * BLKM;
  const int wave = threadIdx.x >> 5;
  const int lane = threadIdx.x & 31;
  const int half = lane >> 4, ln = lane & 15;
  const int nb   = blockIdx.y;                 // 64-col block of C

  const __bf16* arow0 = h_ws + (size_t)(row0 + wave * 32 + ln) * HDIM;
  const __bf16* arow1 = arow0 + (size_t)16 * HDIM;

  constexpr int KT = HDIM / 32;
  const __bf16* bsrc = wpj + ((size_t)e * (CDIM / 64) + nb) * KT * 2048;
  const uint32_t lds0 = (uint32_t)(uintptr_t)&smem[0][0];
  const uint32_t lds1 = (uint32_t)(uintptr_t)&smem[1][0];
  const uint32_t loff = threadIdx.x * 32;

  auto stage = [&](int kt, int buf) {
    const uint64_t g = (uint64_t)(uintptr_t)(bsrc + (size_t)kt * 2048) + loff;
    const uint32_t l = (buf ? lds1 : lds0) + loff;
    async_g2l_b128(l, g);
    async_g2l_b128(l + 16, g + 16);
  };

  f32x8 acc[8] = {};
  stage(0, 0);
  for (int kt = 0; kt < KT; ++kt) {
    const int buf = kt & 1;
    if (kt + 1 < KT) { stage(kt + 1, buf ^ 1); wait_async_le2(); }
    else             { wait_async_0(); }
    __syncthreads();
    const bf16x16 af0 = load_a_frag(arow0, kt * 32, half);
    const bf16x16 af1 = load_a_frag(arow1, kt * 32, half);
#pragma unroll
    for (int t = 0; t < 4; ++t) {
      const bf16x16 b = *(const bf16x16*)&smem[buf][t * 512 + lane * 16];
      acc[t]     = __builtin_amdgcn_wmma_f32_16x16x32_bf16(false, af0, false, b,
                                                           (short)0, acc[t], false, false);
      acc[4 + t] = __builtin_amdgcn_wmma_f32_16x16x32_bf16(false, af1, false, b,
                                                           (short)0, acc[4 + t], false, false);
    }
    __syncthreads();
  }

  int   aid[16];
  float wgt[16];
#pragma unroll
  for (int s = 0; s < 2; ++s)
#pragma unroll
    for (int i = 0; i < 8; ++i) {
      const int slot = row0 + wave * 32 + s * 16 + i + 8 * half;
      aid[s * 8 + i] = sorted_ids[slot];
      wgt[s * 8 + i] = sorted_w[slot];
    }
#pragma unroll
  for (int t = 0; t < 4; ++t) {
    const int ccol = nb * 64 + t * 16 + ln;
    const float bias = bpj[e * CDIM + ccol];
#pragma unroll
    for (int s = 0; s < 2; ++s)
#pragma unroll
      for (int i = 0; i < 8; ++i) {
        if (aid[s * 8 + i] < 0) continue;      // padding slot
        out_partial[(size_t)aid[s * 8 + i] * CDIM + ccol] =
            wgt[s * 8 + i] * (acc[4 * s + t][i] + bias);
      }
  }
}

// ---------------------------------------------------------------------------
// 6) combine the two expert contributions per token in fixed order
// ---------------------------------------------------------------------------
__global__ void k_combine(const float* __restrict__ op, float* __restrict__ y) {
  const int i = blockIdx.x * blockDim.x + threadIdx.x;
  if (i >= NTOK * CDIM) return;
  const int tok = i / CDIM, c = i % CDIM;
  y[i] = op[(size_t)(tok * 2) * CDIM + c] + op[(size_t)(tok * 2 + 1) * CDIM + c];
}

// ---------------------------------------------------------------------------
// launch
// ---------------------------------------------------------------------------
extern "C" void kernel_launch(void* const* d_in, const int* in_sizes, int n_in,
                              void* d_out, int out_size, void* d_ws, size_t ws_size,
                              hipStream_t stream) {
  const float* x    = (const float*)d_in[0];
  const float* rw   = (const float*)d_in[1];
  const float* wfc  = (const float*)d_in[2];
  const float* bfc  = (const float*)d_in[3];
  const float* wpj  = (const float*)d_in[4];
  const float* bpj  = (const float*)d_in[5];
  float* y = (float*)d_out;

  char* ws = (char*)d_ws;
  size_t off = 0;
  auto alloc = [&](size_t bytes) -> void* {
    off = (off + 255) & ~(size_t)255;
    void* p = ws + off;
    off += bytes;
    return p;
  };
  __bf16* xb      = (__bf16*)alloc((size_t)NTOK * CDIM * 2);
  __bf16* wfc_swz = (__bf16*)alloc((size_t)EXP * CDIM * HDIM * 2);
  __bf16* wpj_swz = (__bf16*)alloc((size_t)EXP * HDIM * CDIM * 2);
  __bf16* h_ws    = (__bf16*)alloc((size_t)PADMAX * HDIM * 2);
  float*  op      = (float*)alloc((size_t)NASSIGN * CDIM * 4);
  int*   topk_idx = (int*)alloc((size_t)NASSIGN * 4);
  float* topk_w   = (float*)alloc((size_t)NASSIGN * 4);
  int*   counts   = (int*)alloc(EXP * 4);
  int*   fill     = (int*)alloc(EXP * 4);
  int*   tile_exp = (int*)alloc(MAXT * 4);
  int*   ntiles   = (int*)alloc(4);
  int*   s_ids    = (int*)alloc((size_t)PADMAX * 4);
  float* s_w      = (float*)alloc((size_t)PADMAX * 4);

  // conversions (independent of routing)
  k_cvt_x<<<(NTOK * CDIM / 4 + 255) / 256, 256, 0, stream>>>(x, xb);
  {
    const int frags = EXP * (HDIM / 16) * (CDIM / 32);
    k_cvt_w<<<(frags * 32 + 255) / 256, 256, 0, stream>>>(wfc, wfc_swz, CDIM, HDIM);
  }
  {
    const int frags = EXP * (CDIM / 16) * (HDIM / 32);
    k_cvt_w<<<(frags * 32 + 255) / 256, 256, 0, stream>>>(wpj, wpj_swz, HDIM, CDIM);
  }

  // routing
  k_init<<<(PADMAX + 255) / 256, 256, 0, stream>>>(s_ids, counts);
  k_router<<<NTOK / 8, 256, 0, stream>>>(x, rw, topk_idx, topk_w);
  k_count<<<(NASSIGN + 255) / 256, 256, 0, stream>>>(topk_idx, counts);
  k_meta<<<1, 1, 0, stream>>>(counts, fill, tile_exp, ntiles);
  k_scatter<<<(NASSIGN + 255) / 256, 256, 0, stream>>>(topk_idx, topk_w, fill, s_ids, s_w);

  // expert FFN (grouped GEMMs over sorted assignments)
  dim3 g1(MAXT, HDIM / 64);
  k_gemm1<<<g1, 128, 0, stream>>>(xb, wfc_swz, bfc, s_ids, tile_exp, ntiles, h_ws);
  dim3 g2(MAXT, CDIM / 64);
  k_gemm2<<<g2, 128, 0, stream>>>(h_ws, wpj_swz, bpj, s_ids, s_w, tile_exp, ntiles, op);

  // deterministic combine
  k_combine<<<(NTOK * CDIM + 255) / 256, 256, 0, stream>>>(op, y);
}
